// Gemma4AudioAttention_78400333021385
// MI455X (gfx1250) — matrix operand order repaired and checked by reading, not compiled, not run on hardware
//
#include <hip/hip_runtime.h>
#include <cstdint>
#include <cstddef>

typedef __attribute__((ext_vector_type(16))) _Float16     v16h;
typedef __attribute__((ext_vector_type(8)))  float        v8f;
typedef __attribute__((ext_vector_type(4)))  unsigned int u32x4;
typedef __attribute__((ext_vector_type(4)))  int          i32x4;
typedef __attribute__((ext_vector_type(8)))  int          i32x8;

namespace {

constexpr int HH   = 12;
constexpr int DD   = 128;
constexpr int HID  = 1536;
constexpr int CHK  = 12;
constexpr int BB   = 4;
constexpr int SS   = 6000;
constexpr int MROW = BB * SS;                       // 24000
constexpr int NKST = HID / 32;                      // 48 K-steps
constexpr float QSCALE = 0.12751743f;               // D^-0.5 / ln(2)
constexpr float KSCALE = 1.89463615f;               // ln(1+e) / ln(2)

__device__ __forceinline__ v8f wmma16(v16h a, v16h b, v8f c) {
  return __builtin_amdgcn_wmma_f32_16x16x32_f16(false, a, false, b, (short)0, c,
                                                false, false);
}

// A-matrix fragment (16x32 f16): lane&15 = M row, lane>>4 selects K halves
// {0..7,16..23} vs {8..15,24..31}.  base points at [row0][k0], rs = row stride.
__device__ __forceinline__ v16h fragA(const _Float16* base, int rs, int lane) {
  const int r  = lane & 15;
  const int kh = lane >> 4;
  const _Float16* p = base + r * rs + kh * 8;
  v16h a;
  *reinterpret_cast<float4*>(&a)       = *reinterpret_cast<const float4*>(p);
  *(reinterpret_cast<float4*>(&a) + 1) = *reinterpret_cast<const float4*>(p + 16);
  return a;
}

// B-matrix fragment (32x16 f16): lane&15 = N col, lane>>4 selects K halves
// {0..7,16..23} vs {8..15,24..31}.  base indexed [n][k].
__device__ __forceinline__ v16h fragB(const _Float16* base, int rs, int lane) {
  const int n  = lane & 15;
  const int kb = lane >> 4;
  const _Float16* p = base + n * rs + kb * 8;
  v16h b;
  *reinterpret_cast<float4*>(&b)       = *reinterpret_cast<const float4*>(p);
  *(reinterpret_cast<float4*>(&b) + 1) = *reinterpret_cast<const float4*>(p + 16);
  return b;
}

// TDM: DMA a [rowsTile x 64B] f16 tile (rows of 32 f16) global -> LDS.
// data_size = 8B units; tensor_dim1 = rows remaining (OOB rows read as zero);
// pad_interval = 3 (pad after every 64B) and pad_amount = 3 (16B) give the
// LDS stride-40-f16 layout used by the fragment loaders.
__device__ __forceinline__ void tdm_load_tile(const _Float16* gptr,
                                              unsigned int ldsOff,
                                              unsigned int rowsRem,
                                              unsigned int colsRemUnits) {
  const unsigned long long ga = (unsigned long long)(uintptr_t)gptr;
  u32x4 g0 = {
      1u,                                   // count = 1, user descriptor
      ldsOff,                               // lds_addr (bytes)
      (unsigned int)ga,                     // global_addr[31:0]
      (unsigned int)((ga >> 32) & 0x01ffffffu) | 0x80000000u  // addr[56:32]|type=2
  };
  i32x8 g1 = {
      (int)((3u << 16) | (1u << 20) | (3u << 22) | (3u << 25)),
      //     data_size=8B | pad_en   | pad_int=64B| pad_amt=16B ; wg_mask=0
      (int)((colsRemUnits & 0xffffu) << 16),            // tensor_dim0[15:0]
      (int)(((colsRemUnits >> 16) & 0xffffu) |
            ((rowsRem & 0xffffu) << 16)),               // dim0 hi | dim1 lo
      (int)(((rowsRem >> 16) & 0xffffu) | (8u << 16)),  // dim1 hi | tile_dim0=8
      (int)128,                                         // tile_dim1=128, tile_dim2=0
      (int)384,                                         // tensor_dim0_stride = 1536*2/8
      0, 0                                              // stride hi, dim1_stride = 0
  };
  i32x4 z4 = {0, 0, 0, 0};
#if __clang_major__ >= 23
  i32x8 z8 = {0, 0, 0, 0, 0, 0, 0, 0};
  __builtin_amdgcn_tensor_load_to_lds(g0, g1, z4, z4, z8, 0);
#else
  __builtin_amdgcn_tensor_load_to_lds(g0, g1, z4, z4, 0);
#endif
}

// fp32 -> f16 streaming convert (n4 = count of float4 groups)
__global__ __launch_bounds__(256) void cvt_kernel(const float* __restrict__ s,
                                                  _Float16* __restrict__ d,
                                                  int n4) {
  const int i = blockIdx.x * 256 + threadIdx.x;
  if (i < n4) {
    const float4 f = reinterpret_cast<const float4*>(s)[i];
    _Float16 t[4] = {(_Float16)f.x, (_Float16)f.y, (_Float16)f.z, (_Float16)f.w};
    reinterpret_cast<float2*>(d)[i] = *reinterpret_cast<float2*>(t);
  }
}

// C = A @ W^T with A [M,1536] f16, W [1536,1536] f16 (both k-contiguous).
// Double-buffered TDM staging.  SMODE: 0 none, 1 scalar, 2 per-dim softplus.
template <bool OUTF32, int SMODE>
__global__ __launch_bounds__(256) void gemm_tdm(const _Float16* __restrict__ A,
                                                const _Float16* __restrict__ Wt,
                                                void* __restrict__ Outv, int M,
                                                float sScalar,
                                                const float* __restrict__ pds) {
  __shared__ _Float16 Abuf[2][128 * 40];
  __shared__ _Float16 Bbuf[2][128 * 40];

  const int tid   = threadIdx.x;
  const int lane  = tid & 31;
  const int wave  = tid >> 5;
  const int wm    = wave >> 1;        // 0..3 -> 32-row strip
  const int wn    = wave & 1;         // 0..1 -> 64-col strip
  const int mBase = blockIdx.y * 128;
  const int nBase = blockIdx.x * 128;

  const unsigned int rowsA = (unsigned int)(M - mBase);      // OOB rows -> 0
  const unsigned int rowsB = (unsigned int)(HID - nBase);
  const unsigned int offA[2] = {(unsigned int)(uintptr_t)&Abuf[0][0],
                                (unsigned int)(uintptr_t)&Abuf[1][0]};
  const unsigned int offB[2] = {(unsigned int)(uintptr_t)&Bbuf[0][0],
                                (unsigned int)(uintptr_t)&Bbuf[1][0]};
  const _Float16* aTile = A + (size_t)mBase * HID;
  const _Float16* bTile = Wt + (size_t)nBase * HID;

  if (tid == 0) {  // prologue: stage K-step 0 into buffer 0
    tdm_load_tile(aTile, offA[0], rowsA, (unsigned int)(HID / 4));
    tdm_load_tile(bTile, offB[0], rowsB, (unsigned int)(HID / 4));
  }

  v8f acc[2][4] = {};

  for (int it = 0; it < NKST; ++it) {
    __syncthreads();  // everyone done computing step it-1 -> may overwrite buf[(it+1)&1]
    if (tid == 0) {
      if (it + 1 < NKST) {
        const int kk = (it + 1) * 32;
        tdm_load_tile(aTile + kk, offA[(it + 1) & 1], rowsA,
                      (unsigned int)((HID - kk) / 4));
        tdm_load_tile(bTile + kk, offB[(it + 1) & 1], rowsB,
                      (unsigned int)((HID - kk) / 4));
        __builtin_amdgcn_s_wait_tensorcnt(2);  // in-order: step it complete
      } else {
        __builtin_amdgcn_s_wait_tensorcnt(0);
      }
    }
    __syncthreads();  // buf[it&1] is ready for all waves

    const _Float16* As = Abuf[it & 1];
    const _Float16* Bs = Bbuf[it & 1];
    v16h af[2], bf[4];
#pragma unroll
    for (int mt = 0; mt < 2; ++mt)
      af[mt] = fragA(&As[(wm * 32 + mt * 16) * 40], 40, lane);
#pragma unroll
    for (int nt = 0; nt < 4; ++nt)
      bf[nt] = fragB(&Bs[(wn * 64 + nt * 16) * 40], 40, lane);
#pragma unroll
    for (int mt = 0; mt < 2; ++mt)
#pragma unroll
      for (int nt = 0; nt < 4; ++nt)
        acc[mt][nt] = wmma16(af[mt], bf[nt], acc[mt][nt]);
  }

#pragma unroll
  for (int mt = 0; mt < 2; ++mt) {
#pragma unroll
    for (int nt = 0; nt < 4; ++nt) {
      const int m0 = mBase + wm * 32 + mt * 16 + (lane >> 4) * 8;
      const int n  = nBase + wn * 64 + nt * 16 + (lane & 15);
      float scale = 1.f;
      if constexpr (SMODE == 1) scale = sScalar;
      if constexpr (SMODE == 2) scale = QSCALE * log1pf(expf(pds[n & (DD - 1)]));
#pragma unroll
      for (int r = 0; r < 8; ++r) {
        const int m = m0 + r;
        if (m < M) {
          const float vv = acc[mt][nt][r] * scale;
          if constexpr (OUTF32)
            reinterpret_cast<float*>(Outv)[(size_t)m * HID + n] = vv;
          else
            reinterpret_cast<_Float16*>(Outv)[(size_t)m * HID + n] = (_Float16)vv;
        }
      }
    }
  }
}

// One wave per (batch, head, chunk-block).  S == 500*12 exactly, so no q pad.
__global__ __launch_bounds__(32) void attn_kernel(const _Float16* __restrict__ q,
                                                  const _Float16* __restrict__ k,
                                                  const _Float16* __restrict__ v,
                                                  const _Float16* __restrict__ rel,
                                                  _Float16* __restrict__ out) {
  const int nb   = blockIdx.x;         // 0..499
  const int bh   = blockIdx.y;         // 0..47
  const int b    = bh / HH;
  const int h    = bh % HH;
  const int lane = threadIdx.x;

  __shared__ _Float16 Qs[16 * 136];    // q tile  [16 x 128], stride 136
  __shared__ _Float16 Ks[32 * 136];    // k_ctx   [32 x 128]
  __shared__ _Float16 Rs[32 * 136];    // rel_k   [32 x 128]
  __shared__ _Float16 vT[128 * 40];    // v_ctx^T [128 x 32], stride 40
  __shared__ _Float16 attnS[16 * 40];  // attn    [16 x 32],  stride 40
  __shared__ float    acS[16 * 32];
  __shared__ float    bdS[16 * 32];

  for (int i = lane; i < 128 * 40; i += 32) vT[i] = (_Float16)0.f;
  for (int i = lane; i < 16 * 40; i += 32) attnS[i] = (_Float16)0.f;
  __syncthreads();

  const size_t rowBase =
      ((size_t)b * SS + (size_t)nb * CHK) * (size_t)HID + (size_t)h * DD;

  for (int c = lane; c < 16 * 16; c += 32) {
    const int r = c >> 4, cc = c & 15;
    float4 val;
    if (r < CHK)
      val = *reinterpret_cast<const float4*>(q + rowBase + (size_t)r * HID + cc * 8);
    else
      val = make_float4(0.f, 0.f, 0.f, 0.f);
    *reinterpret_cast<float4*>(&Qs[r * 136 + cc * 8]) = val;
  }
  for (int c = lane; c < 32 * 16; c += 32) {
    const int j = c >> 4, cc = c & 15;
    const int s = nb * CHK + j - 12;
    float4 val = make_float4(0.f, 0.f, 0.f, 0.f);
    if (j < 24 && s >= 0 && s < SS)
      val = *reinterpret_cast<const float4*>(k + ((size_t)b * SS + s) * HID +
                                             h * DD + cc * 8);
    *reinterpret_cast<float4*>(&Ks[j * 136 + cc * 8]) = val;
  }
  for (int c = lane; c < 32 * 16; c += 32) {
    const int p = c >> 4, cc = c & 15;
    float4 val = make_float4(0.f, 0.f, 0.f, 0.f);
    if (p < 24)
      val = *reinterpret_cast<const float4*>(rel + ((size_t)p * HH + h) * DD + cc * 8);
    *reinterpret_cast<float4*>(&Rs[p * 136 + cc * 8]) = val;
  }
  for (int c = lane; c < 24 * 16; c += 32) {
    const int j = c >> 4, cc = c & 15;
    const int s = nb * CHK + j - 12;
    if (s >= 0 && s < SS) {
      _Float16 tmp[8];
      *reinterpret_cast<float4*>(tmp) = *reinterpret_cast<const float4*>(
          v + ((size_t)b * SS + s) * HID + h * DD + cc * 8);
#pragma unroll
      for (int e = 0; e < 8; ++e) vT[(cc * 8 + e) * 40 + j] = tmp[e];
    }
  }
  __syncthreads();

  v8f ac0 = {}, ac1 = {}, bd0 = {}, bd1 = {};
#pragma unroll
  for (int ks = 0; ks < 4; ++ks) {
    const v16h aq = fragA(&Qs[ks * 32], 136, lane);
    ac0 = wmma16(aq, fragB(&Ks[ks * 32], 136, lane), ac0);
    ac1 = wmma16(aq, fragB(&Ks[16 * 136 + ks * 32], 136, lane), ac1);
    bd0 = wmma16(aq, fragB(&Rs[ks * 32], 136, lane), bd0);
    bd1 = wmma16(aq, fragB(&Rs[16 * 136 + ks * 32], 136, lane), bd1);
  }
  {
    const int mlo = (lane >> 4) * 8, n = lane & 15;
#pragma unroll
    for (int r = 0; r < 8; ++r) {
      acS[(mlo + r) * 32 + n]      = ac0[r];
      acS[(mlo + r) * 32 + 16 + n] = ac1[r];
      bdS[(mlo + r) * 32 + n]      = bd0[r];
      bdS[(mlo + r) * 32 + 16 + n] = bd1[r];
    }
  }
  __syncthreads();

  if (lane < CHK) {
    const int qrow = lane;
    float l[24];
    float mx = -1e30f;
#pragma unroll
    for (int j = 0; j < 24; ++j) {
      const int kj   = nb * CHK + j - 12;
      const int dist = qrow - j + 12;
      const bool ok  = (dist >= 0) && (dist < 12) && (kj >= 0) && (kj < SS);
      float val = -1e9f;
      if (ok) {
        const float a_ = acS[qrow * 32 + j];
        const float b_ = bdS[qrow * 32 + (j - qrow)];  // rel-shift: bd[q, j-q]
        val = tanhf((a_ + b_) * 0.02f) * 50.f;
      }
      l[j] = val;
      mx   = fmaxf(mx, val);
    }
    float sum = 0.f;
#pragma unroll
    for (int j = 0; j < 24; ++j) {
      const float e = expf(l[j] - mx);
      l[j] = e;
      sum += e;
    }
    const float inv = 1.f / sum;
#pragma unroll
    for (int j = 0; j < 24; ++j)
      attnS[qrow * 40 + j] = (_Float16)(l[j] * inv);
  }
  __syncthreads();

  const v16h aatt = fragA(&attnS[0], 40, lane);
  const int mlo = (lane >> 4) * 8, n = lane & 15;
#pragma unroll
  for (int tn = 0; tn < 8; ++tn) {
    v8f o = {};
    o = wmma16(aatt, fragB(&vT[tn * 16 * 40], 40, lane), o);
#pragma unroll
    for (int r = 0; r < 8; ++r) {
      const int m = mlo + r;
      if (m < CHK)
        out[rowBase + (size_t)m * HID + tn * 16 + n] = (_Float16)o[r];
    }
  }
}

}  // namespace

extern "C" void kernel_launch(void* const* d_in, const int* in_sizes, int n_in,
                              void* d_out, int out_size, void* d_ws,
                              size_t ws_size, hipStream_t stream) {
  const float* x     = reinterpret_cast<const float*>(d_in[0]);
  const float* pos   = reinterpret_cast<const float*>(d_in[1]);
  const float* Wq    = reinterpret_cast<const float*>(d_in[2]);
  const float* Wk    = reinterpret_cast<const float*>(d_in[3]);
  const float* Wv    = reinterpret_cast<const float*>(d_in[4]);
  const float* Wrel  = reinterpret_cast<const float*>(d_in[5]);
  const float* Wpost = reinterpret_cast<const float*>(d_in[6]);
  const float* pds   = reinterpret_cast<const float*>(d_in[7]);
  float* out         = reinterpret_cast<float*>(d_out);

  _Float16* ws = reinterpret_cast<_Float16*>(d_ws);
  const size_t QE = (size_t)MROW * HID;    // 36,864,000 f16 elems
  const size_t WE = (size_t)HID * HID;     // 2,359,296
  _Float16* q16   = ws;
  _Float16* k16   = ws + QE;
  _Float16* v16   = ws + 2 * QE;
  _Float16* rel16 = ws + 3 * QE;                           // 24*1536
  _Float16* at16  = ws + 3 * QE + (size_t)24 * HID;
  _Float16* x16   = ws + 4 * QE + (size_t)24 * HID;
  _Float16* wq16  = x16 + QE;
  _Float16* wk16  = wq16 + WE;
  _Float16* wv16  = wk16 + WE;
  _Float16* wr16  = wv16 + WE;
  _Float16* wp16  = wr16 + WE;
  _Float16* pos16 = wp16 + WE;                             // 24*1536

  // fp32 -> f16 one-shot conversions
  const int nx4 = (int)(QE / 4), nw4 = (int)(WE / 4), np4 = 24 * HID / 4;
  cvt_kernel<<<(nx4 + 255) / 256, 256, 0, stream>>>(x, x16, nx4);
  cvt_kernel<<<(nw4 + 255) / 256, 256, 0, stream>>>(Wq, wq16, nw4);
  cvt_kernel<<<(nw4 + 255) / 256, 256, 0, stream>>>(Wk, wk16, nw4);
  cvt_kernel<<<(nw4 + 255) / 256, 256, 0, stream>>>(Wv, wv16, nw4);
  cvt_kernel<<<(nw4 + 255) / 256, 256, 0, stream>>>(Wrel, wr16, nw4);
  cvt_kernel<<<(nw4 + 255) / 256, 256, 0, stream>>>(Wpost, wp16, nw4);
  cvt_kernel<<<(np4 + 255) / 256, 256, 0, stream>>>(pos, pos16, np4);

  const dim3 gg(HID / 128, (MROW + 127) / 128);   // 12 x 188
  const dim3 bb(256);

  gemm_tdm<false, 2><<<gg, bb, 0, stream>>>(x16, wq16, q16, MROW, 1.f, pds);
  gemm_tdm<false, 1><<<gg, bb, 0, stream>>>(x16, wk16, k16, MROW, KSCALE, nullptr);
  gemm_tdm<false, 0><<<gg, bb, 0, stream>>>(x16, wv16, v16, MROW, 1.f, nullptr);
  gemm_tdm<false, 0><<<dim3(HID / 128, 1), bb, 0, stream>>>(pos16, wr16, rel16,
                                                            24, 1.f, nullptr);

  attn_kernel<<<dim3(SS / CHK, BB * HH), dim3(32), 0, stream>>>(q16, k16, v16,
                                                                rel16, at16);

  gemm_tdm<true, 0><<<gg, bb, 0, stream>>>(at16, wp16, out, MROW, 1.f, nullptr);
}
